// TEAttention_20091857011280
// MI455X (gfx1250) — matrix operand order repaired and checked by reading, not compiled, not run on hardware
//
#include <hip/hip_runtime.h>
#include <hip/hip_bf16.h>

typedef __attribute__((ext_vector_type(16))) _Float16 v16h;
typedef __attribute__((ext_vector_type(8)))  float    v8f;

#define HW   16384   // 128*128
#define Bsz  16
#define Csz  64
#define EPSI 1e-5f

// ---------------------------------------------------------------------------
// K0: pre-swizzle w_conv1 (64x64 f32) into f16 WMMA A-fragments.
// 8 fragments = (4 row-tiles) x (2 k-steps of 32). Per fragment: 32 lanes x 16 halves.
__global__ void prep_w_kernel(const float* __restrict__ w, _Float16* __restrict__ wf16) {
    int idx = blockIdx.x * blockDim.x + threadIdx.x;   // 0..4095
    if (idx >= 4096) return;
    int f  = idx >> 9;          // fragment 0..7
    int l  = (idx >> 4) & 31;   // lane
    int e  = idx & 15;          // element within v16h
    int mt = f >> 1, ks = f & 1;
    int o  = mt * 16 + (l & 15);
    int hi = l >> 4;
    int kk = ks * 32 + ((e >> 3) * 16 + hi * 8 + (e & 7));
    wf16[idx] = (_Float16)w[o * 64 + kk];
}

// ---------------------------------------------------------------------------
// K1: conv1x1 via WMMA f16->f32, fused with BN stage 0 of the PR branch.
__global__ void conv1_wmma_kernel(const float* __restrict__ x,
                                  const _Float16* __restrict__ wf16,
                                  const float* __restrict__ pr_gamma0,
                                  const float* __restrict__ pr_beta0,
                                  float* __restrict__ out) {
    const int lane = threadIdx.x & 31;
    const int wave = threadIdx.x >> 5;
    const int colBase = blockIdx.x * 128 + wave * 16;
    const int b   = colBase >> 14;
    const int pix = colBase & (HW - 1);
    const int nlo = lane & 15;
    const int hi  = lane >> 4;

    v16h a[4][2];
#pragma unroll
    for (int mt = 0; mt < 4; ++mt)
#pragma unroll
        for (int ks = 0; ks < 2; ++ks)
            a[mt][ks] = *reinterpret_cast<const v16h*>(wf16 + ((mt * 2 + ks) * 32 + lane) * 16);

    v8f acc0 = {}, acc1 = {}, acc2 = {}, acc3 = {};
#pragma unroll
    for (int ks = 0; ks < 2; ++ks) {
        v16h bf;
#pragma unroll
        for (int e = 0; e < 16; ++e) {
            int c = ks * 32 + ((e >> 3) * 16 + hi * 8 + (e & 7));
            bf[e] = (_Float16)x[((b * Csz + c) << 14) + pix + nlo];
        }
        acc0 = __builtin_amdgcn_wmma_f32_16x16x32_f16(false, a[0][ks], false, bf, (short)0, acc0, false, false);
        acc1 = __builtin_amdgcn_wmma_f32_16x16x32_f16(false, a[1][ks], false, bf, (short)0, acc1, false, false);
        acc2 = __builtin_amdgcn_wmma_f32_16x16x32_f16(false, a[2][ks], false, bf, (short)0, acc2, false, false);
        acc3 = __builtin_amdgcn_wmma_f32_16x16x32_f16(false, a[3][ks], false, bf, (short)0, acc3, false, false);
    }

    const float inv = rsqrtf(1.0f + EPSI);
    v8f accs[4] = {acc0, acc1, acc2, acc3};
#pragma unroll
    for (int mt = 0; mt < 4; ++mt)
#pragma unroll
        for (int r = 0; r < 8; ++r) {
            int o = mt * 16 + r + hi * 8;
            float v = accs[mt][r] * (pr_gamma0[o] * inv) + pr_beta0[o];
            out[((b * Csz + o) << 14) + pix + nlo] = v;
        }
}

// ---------------------------------------------------------------------------
// K2: fused 4-stage separable depthwise chain + BN stages 1..4, LDS-tiled.
// 32x32 output tile, 44x44 input halo (+-6).  Out-of-image positions are
// forced to 0 at every stage so each conv sees correct zero padding.
#define TS   32
#define TH_  44
__global__ void dw_chain_kernel(const float* __restrict__ in, float* __restrict__ out,
                                const float* __restrict__ pr_w1, const float* __restrict__ pr_w2,
                                const float* __restrict__ pr_w3, const float* __restrict__ pr_w4,
                                const float* __restrict__ pr_gamma, const float* __restrict__ pr_beta) {
    __shared__ float t0[TH_ * TH_];
    __shared__ float t1[TH_ * TH_];
    const int tid  = threadIdx.x;
    const int bc   = blockIdx.x >> 4;        // b*64 + c
    const int tile = blockIdx.x & 15;
    const int c    = bc & 63;
    const int oh   = (tile >> 2) * TS;       // tile origin in image
    const int ow   = (tile & 3) * TS;
    const float inv = rsqrtf(1.0f + EPSI);

    float w1[5], w2[5], w3[5], w4[5];
#pragma unroll
    for (int t = 0; t < 5; ++t) {
        w1[t] = pr_w1[c * 5 + t]; w2[t] = pr_w2[c * 5 + t];
        w3[t] = pr_w3[c * 5 + t]; w4[t] = pr_w4[c * 5 + t];
    }
    const float g1 = pr_gamma[64  + c] * inv, b1 = pr_beta[64  + c];
    const float g2 = pr_gamma[128 + c] * inv, b2 = pr_beta[128 + c];
    const float g3 = pr_gamma[192 + c] * inv, b3 = pr_beta[192 + c];
    const float g4 = pr_gamma[256 + c] * inv, b4 = pr_beta[256 + c];
    const float* plane = in + ((long)bc << 14);

    // load 44x44 halo (zeros outside image)
    for (int idx = tid; idx < TH_ * TH_; idx += 256) {
        int r = idx / TH_, cc = idx % TH_;
        int gh = oh + r - 6, gw = ow + cc - 6;
        t0[idx] = (gh >= 0 && gh < 128 && gw >= 0 && gw < 128) ? plane[gh * 128 + gw] : 0.0f;
    }
    __syncthreads();

    // stage 1: horizontal +-2, dil 1  -> t1, region rows[0,44) cols[2,42)
    for (int idx = tid; idx < 44 * 40; idx += 256) {
        int r = idx / 40, cc = 2 + idx % 40;
        int gh = oh + r - 6, gw = ow + cc - 6;
        float v = 0.0f;
        if (gh >= 0 && gh < 128 && gw >= 0 && gw < 128) {
            float s = 0.0f;
#pragma unroll
            for (int t = 0; t < 5; ++t) s += t0[r * TH_ + cc - 2 + t] * w1[t];
            v = s * g1 + b1;
        }
        t1[r * TH_ + cc] = v;
    }
    __syncthreads();

    // stage 2: vertical +-2, dil 1  -> t0, region rows[2,42) cols[2,42)
    for (int idx = tid; idx < 40 * 40; idx += 256) {
        int r = 2 + idx / 40, cc = 2 + idx % 40;
        int gh = oh + r - 6, gw = ow + cc - 6;
        float v = 0.0f;
        if (gh >= 0 && gh < 128 && gw >= 0 && gw < 128) {
            float s = 0.0f;
#pragma unroll
            for (int t = 0; t < 5; ++t) s += t1[(r - 2 + t) * TH_ + cc] * w2[t];
            v = s * g2 + b2;
        }
        t0[r * TH_ + cc] = v;
    }
    __syncthreads();

    // stage 3: horizontal +-4, dil 2 -> t1, region rows[2,42) cols[6,38)
    for (int idx = tid; idx < 40 * 32; idx += 256) {
        int r = 2 + idx / 32, cc = 6 + idx % 32;
        int gh = oh + r - 6, gw = ow + cc - 6;
        float v = 0.0f;
        if (gh >= 0 && gh < 128 && gw >= 0 && gw < 128) {
            float s = 0.0f;
#pragma unroll
            for (int t = 0; t < 5; ++t) s += t0[r * TH_ + cc - 4 + 2 * t] * w3[t];
            v = s * g3 + b3;
        }
        t1[r * TH_ + cc] = v;
    }
    __syncthreads();

    // stage 4: vertical +-4, dil 2 -> global, region rows[6,38) cols[6,38)
    float* oplane = out + ((long)bc << 14);
    for (int idx = tid; idx < 32 * 32; idx += 256) {
        int r = 6 + idx / 32, cc = 6 + idx % 32;
        float s = 0.0f;
#pragma unroll
        for (int t = 0; t < 5; ++t) s += t1[(r - 4 + 2 * t) * TH_ + cc] * w4[t];
        int gh = oh + r - 6, gw = ow + cc - 6;
        oplane[gh * 128 + gw] = s * g4 + b4;
    }
}

// ---------------------------------------------------------------------------
// K3: g[b,c] = mean over HW of x.
__global__ void gmean_kernel(const float* __restrict__ x, float* __restrict__ g) {
    __shared__ float red[256];
    int bc = blockIdx.x;
    const float* p = x + ((long)bc << 14);
    float s = 0.0f;
    for (int i = threadIdx.x; i < HW; i += 256) s += p[i];
    red[threadIdx.x] = s;
    __syncthreads();
    for (int k = 128; k > 0; k >>= 1) {
        if (threadIdx.x < k) red[threadIdx.x] += red[threadIdx.x + k];
        __syncthreads();
    }
    if (threadIdx.x == 0) g[bc] = red[0] * (1.0f / (float)HW);
}

// ---------------------------------------------------------------------------
// K4: cosine similarity map S[b,k].
__global__ void cos_s_kernel(const float* __restrict__ x, const float* __restrict__ g,
                             float* __restrict__ S) {
    __shared__ float gl[64];
    __shared__ float gns;
    int b = blockIdx.x >> 6;
    int k = ((blockIdx.x & 63) << 8) + threadIdx.x;
    if (threadIdx.x < 64) gl[threadIdx.x] = g[b * 64 + threadIdx.x];
    __syncthreads();
    if (threadIdx.x == 0) {
        float s2 = 0.0f;
        for (int c = 0; c < 64; ++c) s2 += gl[c] * gl[c];
        gns = fmaxf(sqrtf(s2), 1e-8f);
    }
    __syncthreads();
    const float* xb = x + ((long)(b * Csz) << 14) + k;
    float num = 0.0f, xs = 0.0f;
    for (int c = 0; c < 64; ++c) {
        float xv = xb[c << 14];
        num += gl[c] * xv;
        xs  += xv * xv;
    }
    float xn = fmaxf(sqrtf(xs), 1e-8f);
    S[(b << 14) + k] = num / (gns * xn);
}

// ---------------------------------------------------------------------------
// K5: per-batch min/max of S.
__global__ void minmax_kernel(const float* __restrict__ S, float* __restrict__ mnmx) {
    __shared__ float rmn[256], rmx[256];
    int b = blockIdx.x;
    const float* p = S + (b << 14);
    float mn = 3.4e38f, mx = -3.4e38f;
    for (int i = threadIdx.x; i < HW; i += 256) {
        float v = p[i];
        mn = fminf(mn, v); mx = fmaxf(mx, v);
    }
    rmn[threadIdx.x] = mn; rmx[threadIdx.x] = mx;
    __syncthreads();
    for (int k = 128; k > 0; k >>= 1) {
        if (threadIdx.x < k) {
            rmn[threadIdx.x] = fminf(rmn[threadIdx.x], rmn[threadIdx.x + k]);
            rmx[threadIdx.x] = fmaxf(rmx[threadIdx.x], rmx[threadIdx.x + k]);
        }
        __syncthreads();
    }
    if (threadIdx.x == 0) { mnmx[b * 2] = rmn[0]; mnmx[b * 2 + 1] = rmx[0]; }
}

// ---------------------------------------------------------------------------
// K6: soft-bin membership V, stored pixel-major [b][k*10+m] (the torch raw
// view Vr[b,m,k] is then just flat offset m*HW+k into the same buffer).
__global__ void mkV_kernel(const float* __restrict__ S, const float* __restrict__ mnmx,
                           float* __restrict__ V) {
    int b = blockIdx.x >> 6;
    int k = ((blockIdx.x & 63) << 8) + threadIdx.x;
    float s  = S[(b << 14) + k];
    float mn = mnmx[b * 2], mx = mnmx[b * 2 + 1];
    float* vp = V + (long)b * (HW * 10) + (long)k * 10;
#pragma unroll
    for (int m = 0; m < 10; ++m) {
        float lv = mn + (mx - mn) * ((float)m * (1.0f / 9.0f));
        float d  = fabsf(lv - s);
        vp[m] = (d < 0.05f) ? (1.0f - d) : 0.0f;
    }
}

// K7: colsum[b,m] = sum_k V[b,k,m].
__global__ void colsum_kernel(const float* __restrict__ V, float* __restrict__ colsum) {
    __shared__ float red[256];
    int b = blockIdx.x / 10, m = blockIdx.x % 10;
    const float* p = V + (long)b * (HW * 10) + m;
    float s = 0.0f;
    for (int i = threadIdx.x; i < HW; i += 256) s += p[(long)i * 10];
    red[threadIdx.x] = s;
    __syncthreads();
    for (int k = 128; k > 0; k >>= 1) {
        if (threadIdx.x < k) red[threadIdx.x] += red[threadIdx.x + k];
        __syncthreads();
    }
    if (threadIdx.x == 0) colsum[b * 10 + m] = red[0];
}

// ---------------------------------------------------------------------------
// K8: tiny per-batch head: cumsum-normalized Ch -> fc1 (raw-view Cq),
// phi1/2/3, 10x10 softmax, Lp = p3 @ X.
__global__ void attn_small_kernel(const float* __restrict__ colsum, const float* __restrict__ mnmx,
                                  const float* __restrict__ fc1_w, const float* __restrict__ fc1_b,
                                  const float* __restrict__ phi1_w, const float* __restrict__ phi1_b,
                                  const float* __restrict__ phi2_w, const float* __restrict__ phi2_b,
                                  const float* __restrict__ phi3_w, const float* __restrict__ phi3_b,
                                  float* __restrict__ Lp) {
    __shared__ float cq[640], q1[640], q2[640], q3[640], xm[100], ch0[10], lv[10];
    int b = blockIdx.x, tid = threadIdx.x;
    if (tid == 0) {
        float mn = mnmx[b * 2], mx = mnmx[b * 2 + 1];
        float acc = 0.0f;
        for (int m = 0; m < 10; ++m) {
            acc += colsum[b * 10 + m];
            ch0[m] = colsum[b * 10 + m] / acc;
            lv[m]  = mn + (mx - mn) * ((float)m * (1.0f / 9.0f));
        }
    }
    __syncthreads();
    for (int idx = tid; idx < 640; idx += 256) {
        int row = idx >> 6, col = idx & 63;
        cq[idx] = ch0[row] * fc1_w[col * 2 + 0] + lv[row] * fc1_w[col * 2 + 1] + fc1_b[col];
    }
    __syncthreads();
    for (int idx = tid; idx < 640; idx += 256) {
        int o = idx / 10, m = idx % 10;
        float s1 = 0.0f, s2 = 0.0f, s3 = 0.0f;
        for (int c = 0; c < 64; ++c) {
            float cv = cq[c * 10 + m];
            s1 += phi1_w[o * 64 + c] * cv;
            s2 += phi2_w[o * 64 + c] * cv;
            s3 += phi3_w[o * 64 + c] * cv;
        }
        q1[idx] = s1 + phi1_b[o];
        q2[idx] = s2 + phi2_b[o];
        q3[idx] = s3 + phi3_b[o];
    }
    __syncthreads();
    if (tid < 100) {
        int m = tid / 10, n = tid % 10;
        float s = 0.0f;
        for (int c = 0; c < 64; ++c) s += q1[c * 10 + m] * q2[c * 10 + n];
        xm[tid] = s;
    }
    __syncthreads();
    if (tid < 10) {
        float mx = -3.4e38f;
        for (int n = 0; n < 10; ++n) mx = fmaxf(mx, xm[tid * 10 + n]);
        float s = 0.0f;
        for (int n = 0; n < 10; ++n) { float e = __expf(xm[tid * 10 + n] - mx); xm[tid * 10 + n] = e; s += e; }
        float r = 1.0f / s;
        for (int n = 0; n < 10; ++n) xm[tid * 10 + n] *= r;
    }
    __syncthreads();
    for (int idx = tid; idx < 640; idx += 256) {
        int c = idx / 10, n = idx % 10;
        float s = 0.0f;
        for (int m = 0; m < 10; ++m) s += q3[c * 10 + m] * xm[m * 10 + n];
        Lp[b * 640 + idx] = s;
    }
}

// ---------------------------------------------------------------------------
// K9: fused R = Lp@Vr, BN1, BN2(w_conv2*S), +x_center, sigmoid, *x.
__global__ void final_fuse_kernel(const float* __restrict__ x, const float* __restrict__ xcen,
                                  const float* __restrict__ S, const float* __restrict__ V,
                                  const float* __restrict__ Lp,
                                  const float* __restrict__ bn1_g, const float* __restrict__ bn1_b,
                                  const float* __restrict__ bn2_g, const float* __restrict__ bn2_b,
                                  const float* __restrict__ w_conv2,
                                  float* __restrict__ out) {
    __shared__ float lp[640], s1[64], o1[64], s2[64], o2[64], wc[64];
    int b = blockIdx.x >> 6;
    int k = ((blockIdx.x & 63) << 8) + threadIdx.x;
    int tid = threadIdx.x;
    for (int i = tid; i < 640; i += 256) lp[i] = Lp[b * 640 + i];
    if (tid < 64) {
        float inv = rsqrtf(1.0f + EPSI);
        s1[tid] = bn1_g[tid] * inv; o1[tid] = bn1_b[tid];
        s2[tid] = bn2_g[tid] * inv; o2[tid] = bn2_b[tid];
        wc[tid] = w_conv2[tid];
    }
    __syncthreads();
    float sv = S[(b << 14) + k];
    float v[10];
#pragma unroll
    for (int m = 0; m < 10; ++m)
        v[m] = V[(long)b * (HW * 10) + (m << 14) + k];
    const long basebk = (long)(b * Csz) << 14;
    for (int c = 0; c < 64; ++c) {
        float r = 0.0f;
#pragma unroll
        for (int m = 0; m < 10; ++m) r += lp[c * 10 + m] * v[m];
        float xt = r * s1[c] + o1[c];
        float xf = wc[c] * sv * s2[c] + o2[c];
        long idx = basebk + ((long)c << 14) + k;
        float a = xf + xt + xcen[idx];
        float att = 1.0f / (1.0f + __expf(-a));
        out[idx] = x[idx] * att;
    }
}

// ---------------------------------------------------------------------------
extern "C" void kernel_launch(void* const* d_in, const int* in_sizes, int n_in,
                              void* d_out, int out_size, void* d_ws, size_t ws_size,
                              hipStream_t stream) {
    (void)in_sizes; (void)n_in; (void)out_size; (void)ws_size;
    const float* x        = (const float*)d_in[0];
    const float* w_conv1  = (const float*)d_in[1];
    const float* pr_w1    = (const float*)d_in[2];
    const float* pr_w2    = (const float*)d_in[3];
    const float* pr_w3    = (const float*)d_in[4];
    const float* pr_w4    = (const float*)d_in[5];
    const float* pr_gamma = (const float*)d_in[6];   // [5,64]
    const float* pr_beta  = (const float*)d_in[7];
    const float* bn1_g    = (const float*)d_in[8];
    const float* bn1_b    = (const float*)d_in[9];
    const float* bn2_g    = (const float*)d_in[10];
    const float* bn2_b    = (const float*)d_in[11];
    const float* fc1_w    = (const float*)d_in[12];
    const float* fc1_b    = (const float*)d_in[13];
    const float* phi1_w   = (const float*)d_in[14];
    const float* phi1_b   = (const float*)d_in[15];
    const float* phi2_w   = (const float*)d_in[16];
    const float* phi2_b   = (const float*)d_in[17];
    const float* phi3_w   = (const float*)d_in[18];
    const float* phi3_b   = (const float*)d_in[19];
    const float* w_conv2  = (const float*)d_in[20];
    float* out = (float*)d_out;

    char* ws = (char*)d_ws;
    const size_t PLANE = (size_t)Bsz * Csz * HW * 4;             // 67,108,864 B
    float*     buf0   = (float*)(ws);                            // conv1 out
    float*     buf1   = (float*)(ws + PLANE);                    // x_center
    float*     Sbuf   = (float*)(ws + 2 * PLANE);                // 1 MB
    float*     Vbuf   = (float*)(ws + 2 * PLANE + 1048576);      // 10.5 MB
    float*     gbuf   = (float*)(ws + 2 * PLANE + 11534336);
    float*     mnmx   = (float*)(ws + 2 * PLANE + 11538432);
    float*     colsum = (float*)(ws + 2 * PLANE + 11538688);
    float*     Lpbuf  = (float*)(ws + 2 * PLANE + 11539712);
    _Float16*  wf16   = (_Float16*)(ws + 2 * PLANE + 11581440);  // 8 KB, 256-aligned

    prep_w_kernel<<<16, 256, 0, stream>>>(w_conv1, wf16);

    // conv1x1 (WMMA) + BN stage 0  -> buf0
    conv1_wmma_kernel<<<(Bsz * HW) / 128, 256, 0, stream>>>(x, wf16, pr_gamma + 0, pr_beta + 0, buf0);

    // fused depthwise chain (BN stages 1..4)  buf0 -> buf1 (x_center)
    dw_chain_kernel<<<Bsz * Csz * 16, 256, 0, stream>>>(buf0, buf1, pr_w1, pr_w2, pr_w3, pr_w4,
                                                        pr_gamma, pr_beta);

    gmean_kernel<<<Bsz * Csz, 256, 0, stream>>>(x, gbuf);
    cos_s_kernel<<<(Bsz * HW) / 256, 256, 0, stream>>>(x, gbuf, Sbuf);
    minmax_kernel<<<Bsz, 256, 0, stream>>>(Sbuf, mnmx);
    mkV_kernel<<<(Bsz * HW) / 256, 256, 0, stream>>>(Sbuf, mnmx, Vbuf);
    colsum_kernel<<<Bsz * 10, 256, 0, stream>>>(Vbuf, colsum);
    attn_small_kernel<<<Bsz, 256, 0, stream>>>(colsum, mnmx, fc1_w, fc1_b,
                                               phi1_w, phi1_b, phi2_w, phi2_b,
                                               phi3_w, phi3_b, Lpbuf);
    final_fuse_kernel<<<(Bsz * HW) / 256, 256, 0, stream>>>(x, buf1, Sbuf, Vbuf, Lpbuf,
                                                            bn1_g, bn1_b, bn2_g, bn2_b,
                                                            w_conv2, out);
}
